// CustomMultiHeadSelfAttention_32925219291536
// MI455X (gfx1250) — matrix operand order repaired and checked by reading, not compiled, not run on hardware
//
#include <hip/hip_runtime.h>
#include <hip/hip_bf16.h>

// MHA: qkv[4,2048,1024] -> proj(QKV) -> 16-head attention (hd=64) -> out proj.
// Strategy: bf16 WMMA (v_wmma_f32_16x16x32_bf16) everywhere, f32 accumulate.
// GEMMs: block-cooperative async global->LDS staging of the weight tile
// (double-buffered, ASYNCcnt) via inline asm global_load_async_to_lds_b128.
// Workspace layout (needs 64 MB):
//   [ 0MB) qkv_bf   bf16 [8192,1024]
//   [16MB) WinT     bf16 [3072,1024]   (W_in transposed, N-major)
//   [22MB) WoutT    bf16 [1024,1024]   (W_out transposed, N-major)
//   [24MB) Qb       bf16 [B*H, 2048, 64]
//   [32MB) Kb       bf16 [B*H, 2048, 64]
//   [40MB) VTb      bf16 [B*H, 64, 2048]   (V transposed per head)
//   [48MB) ctx      bf16 [8192,1024]

typedef __attribute__((ext_vector_type(8)))  __bf16 bf16x8;
typedef __attribute__((ext_vector_type(16))) __bf16 bf16x16;
typedef __attribute__((ext_vector_type(8)))  float  f32x8;

#if defined(__gfx1250__)
#define HAVE_ASYNC_LDS 1
#else
#define HAVE_ASYNC_LDS 0
#endif

#if HAVE_ASYNC_LDS
// Async DMA: 16B per lane, global -> LDS, tracked by ASYNCcnt.
// VDST = per-lane LDS byte address (low 32 bits of a generic LDS pointer are
// the LDS address per the flat-aperture rules); VADDR = 64-bit global address.
__device__ __forceinline__ void async_copy16(const void* g, void* l) {
  unsigned lds = (unsigned)(uintptr_t)l;
  unsigned long long ga = (unsigned long long)(uintptr_t)g;
  asm volatile("global_load_async_to_lds_b128 %0, %1, off"
               :
               : "v"(lds), "v"(ga)
               : "memory");
}
__device__ __forceinline__ void wait_async0() {
#if __has_builtin(__builtin_amdgcn_s_wait_asynccnt)
  __builtin_amdgcn_s_wait_asynccnt(0);
#else
  asm volatile("s_wait_asynccnt 0" ::: "memory");
#endif
}
#endif

__device__ __forceinline__ f32x8 zero8() {
  f32x8 z = {0.f, 0.f, 0.f, 0.f, 0.f, 0.f, 0.f, 0.f};
  return z;
}

__device__ __forceinline__ bf16x16 frag_cat(bf16x8 lo, bf16x8 hi) {
  bf16x16 r;
#pragma unroll
  for (int i = 0; i < 8; ++i) { r[i] = lo[i]; r[i + 8] = hi[i]; }
  return r;
}

// A fragment (16M x 32K) from row-major bf16 src (leading dim ld elements).
// ISA layout: lanes 0-15 hold row M=lane, K{0..7} in V0-3, K{16..23} in V4-7;
//             lanes 16-31 hold row M=lane-16, K{8..15}, K{24..31}.
__device__ __forceinline__ bf16x16 load_a_frag(const __bf16* src, int ld,
                                               int k0, int lane) {
  int r = lane & 15, h = lane >> 4;
  const __bf16* p = src + r * ld + k0 + h * 8;
  bf16x8 lo = *(const bf16x8*)p;
  bf16x8 hi = *(const bf16x8*)(p + 16);
  return frag_cat(lo, hi);
}

// B fragment (32K x 16N) from N-major bf16 src (one row per N, contraction
// contiguous, leading dim ld). ISA layout: lanes 0-15 hold col N=lane,
// K{0..7},K{16..23}; lanes 16-31 hold col N=lane-16, K{8..15},K{24..31}.
__device__ __forceinline__ bf16x16 load_b_frag(const __bf16* src, int ld,
                                               int k0, int lane) {
  int n = lane & 15, h = lane >> 4;
  const __bf16* p = src + n * ld + k0 + h * 8;
  bf16x8 lo = *(const bf16x8*)p;
  bf16x8 hi = *(const bf16x8*)(p + 16);
  return frag_cat(lo, hi);
}

__device__ __forceinline__ f32x8 wmma_bf16(bf16x16 a, bf16x16 b, f32x8 c) {
  return __builtin_amdgcn_wmma_f32_16x16x32_bf16(
      false, a, false, b, (short)0, c, false, false);
}

// ---------------------------------------------------------------- converts
__global__ __launch_bounds__(256) void cvt_bf16_kernel(const float* __restrict__ in,
                                                       __bf16* __restrict__ out,
                                                       int n) {
  for (int i = blockIdx.x * blockDim.x + threadIdx.x; i < n;
       i += gridDim.x * blockDim.x)
    out[i] = (__bf16)in[i];
}

// out[n*1024 + k] = (bf16) in[k*N + n]   (K = 1024 fixed)
__global__ __launch_bounds__(256) void transpose_bf16_kernel(
    const float* __restrict__ in, __bf16* __restrict__ out, int N, int total) {
  for (int idx = blockIdx.x * blockDim.x + threadIdx.x; idx < total;
       idx += gridDim.x * blockDim.x) {
    int n = idx >> 10;
    int k = idx & 1023;
    out[idx] = (__bf16)in[k * N + n];
  }
}

// ---------------------------------------------------------------- QKV GEMM
// C[8192,3072] = qkv_bf @ W_in + b_in, scattered into Q/K/VT head layouts.
// grid (48 Ntiles, 64 Mtiles), 128 threads (4 waves); wave owns 32M x 64N.
// Weight tile (64N x 32K, 4KB) staged once per block via async->LDS DMA,
// double buffered; A rows stream directly from global.
__global__ __launch_bounds__(128) void qkv_proj_kernel(
    const __bf16* __restrict__ A, const __bf16* __restrict__ WT,
    const float* __restrict__ bias, __bf16* __restrict__ Qb,
    __bf16* __restrict__ Kb, __bf16* __restrict__ VTb) {
  const int tid = threadIdx.x;
  const int lane = tid & 31, wave = tid >> 5;
  const int row0 = blockIdx.y * 128 + wave * 32;
  const int nbase = blockIdx.x * 64;
  const int half = lane >> 4, col = lane & 15;

  f32x8 acc[2][4];
#pragma unroll
  for (int s = 0; s < 2; ++s)
#pragma unroll
    for (int nt = 0; nt < 4; ++nt) acc[s][nt] = zero8();

  const __bf16* Arow0 = A + row0 * 1024;
  const __bf16* Arow1 = A + (row0 + 16) * 1024;

#if HAVE_ASYNC_LDS
  __shared__ __align__(16) __bf16 Bl[2][64][48];  // 48 = 32 + 16 pad (banks)
  // tile = 64 rows x 64B = 256 x 16B chunks; 2 chunks per thread
  auto stage = [&](int k0, int buf) {
#pragma unroll
    for (int u = 0; u < 2; ++u) {
      const int c = tid + u * 128;
      const int rw = c >> 2, part = c & 3;
      async_copy16(WT + (nbase + rw) * 1024 + k0 + part * 8,
                   &Bl[buf][rw][part * 8]);
    }
  };
  stage(0, 0);
  wait_async0();
  __syncthreads();
  int buf = 0;
  for (int k0 = 0; k0 < 1024; k0 += 32) {
    if (k0 + 32 < 1024) stage(k0 + 32, buf ^ 1);
    __builtin_prefetch(Arow0 + (lane & 15) * 1024 + k0 + 128, 0, 1);
    bf16x16 a0 = load_a_frag(Arow0, 1024, k0, lane);
    bf16x16 a1 = load_a_frag(Arow1, 1024, k0, lane);
#pragma unroll
    for (int nt = 0; nt < 4; ++nt) {
      bf16x16 b = load_b_frag(&Bl[buf][nt * 16][0], 48, 0, lane);
      acc[0][nt] = wmma_bf16(a0, b, acc[0][nt]);
      acc[1][nt] = wmma_bf16(a1, b, acc[1][nt]);
    }
    wait_async0();     // next tile resident
    __syncthreads();   // all waves done reading current tile
    buf ^= 1;
  }
#else
#pragma unroll 2
  for (int k0 = 0; k0 < 1024; k0 += 32) {
    __builtin_prefetch(Arow0 + (lane & 15) * 1024 + k0 + 128, 0, 1);
    bf16x16 a0 = load_a_frag(Arow0, 1024, k0, lane);
    bf16x16 a1 = load_a_frag(Arow1, 1024, k0, lane);
#pragma unroll
    for (int nt = 0; nt < 4; ++nt) {
      bf16x16 b = load_b_frag(WT + (nbase + nt * 16) * 1024, 1024, k0, lane);
      acc[0][nt] = wmma_bf16(a0, b, acc[0][nt]);
      acc[1][nt] = wmma_bf16(a1, b, acc[1][nt]);
    }
  }
#endif

#pragma unroll
  for (int sub = 0; sub < 2; ++sub) {
#pragma unroll
    for (int nt = 0; nt < 4; ++nt) {
      const int gcol = nbase + nt * 16 + col;
      const float bv = bias[gcol];
      const int which = gcol >> 10;       // 0=Q 1=K 2=V (uniform per nt)
      const int d = gcol & 1023;
      const int h = d >> 6, di = d & 63;
#pragma unroll
      for (int i = 0; i < 8; ++i) {
        const float v = acc[sub][nt][i] + bv;
        const int grow = row0 + sub * 16 + i + 8 * half;  // [0, 8192)
        const int b = grow >> 11, s = grow & 2047;        // batch, seq
        const int bh = b * 16 + h;
        if (which == 0)
          Qb[(bh * 2048 + s) * 64 + di] = (__bf16)v;
        else if (which == 1)
          Kb[(bh * 2048 + s) * 64 + di] = (__bf16)v;
        else
          VTb[(bh * 64 + di) * 2048 + s] = (__bf16)v;
      }
    }
  }
}

// ---------------------------------------------------------------- attention
// Flash attention. grid (64 heads, 16 qtiles), 128 threads (4 waves);
// each wave owns 32 queries (two 16-row sub-tiles). K/V fragments are loaded
// once per 32-key tile and reused by both sub-tiles -> 1.0 load/WMMA.
__global__ __launch_bounds__(128) void attention_kernel(
    const __bf16* __restrict__ Qb, const __bf16* __restrict__ Kb,
    const __bf16* __restrict__ VTb, __bf16* __restrict__ ctx) {
  __shared__ __align__(16) __bf16 Plds[4][2][16][32];  // per-wave P tiles

  const int lane = threadIdx.x & 31, wave = threadIdx.x >> 5;
  const int half = lane >> 4, col = lane & 15;
  const int bh = blockIdx.x;                  // b*16 + h
  const int q0 = blockIdx.y * 128 + wave * 32;

  const __bf16* Qh = Qb + bh * 2048 * 64;
  const __bf16* Kh = Kb + bh * 2048 * 64;
  const __bf16* Vh = VTb + bh * 64 * 2048;

  bf16x16 qa[2][2];
#pragma unroll
  for (int sub = 0; sub < 2; ++sub)
#pragma unroll
    for (int kp = 0; kp < 2; ++kp)
      qa[sub][kp] = load_a_frag(Qh + (q0 + sub * 16) * 64, 64, kp * 32, lane);

  float m[2][8], l[2][8];
  f32x8 o[2][4];
#pragma unroll
  for (int sub = 0; sub < 2; ++sub) {
#pragma unroll
    for (int i = 0; i < 8; ++i) { m[sub][i] = -3.0e38f; l[sub][i] = 0.f; }
#pragma unroll
    for (int np = 0; np < 4; ++np) o[sub][np] = zero8();
  }

  // exp2-domain softmax: fold 1/sqrt(64) * log2(e) into one scale; exp2f
  // lowers directly to v_exp_f32.
  const float SC = 0.125f * 1.44269504088896f;

  for (int kb = 0; kb < 2048; kb += 32) {
    bf16x16 kf[2][2];
#pragma unroll
    for (int nt = 0; nt < 2; ++nt)
#pragma unroll
      for (int kp = 0; kp < 2; ++kp)
        kf[nt][kp] = load_b_frag(Kh + (kb + nt * 16) * 64, 64, kp * 32, lane);

#pragma unroll
    for (int sub = 0; sub < 2; ++sub) {
      f32x8 s0 = zero8(), s1 = zero8();
      s0 = wmma_bf16(qa[sub][0], kf[0][0], s0);
      s0 = wmma_bf16(qa[sub][1], kf[0][1], s0);
      s1 = wmma_bf16(qa[sub][0], kf[1][0], s1);
      s1 = wmma_bf16(qa[sub][1], kf[1][1], s1);

#pragma unroll
      for (int i = 0; i < 8; ++i) {
        const float a0 = s0[i] * SC;
        const float a1 = s1[i] * SC;
        float t = fmaxf(a0, a1);          // row max across the 16-lane half
        t = fmaxf(t, __shfl_xor(t, 8, 32));
        t = fmaxf(t, __shfl_xor(t, 4, 32));
        t = fmaxf(t, __shfl_xor(t, 2, 32));
        t = fmaxf(t, __shfl_xor(t, 1, 32));
        const float mn = fmaxf(m[sub][i], t);
        const float p0 = exp2f(a0 - mn);
        const float p1 = exp2f(a1 - mn);
        float rs = p0 + p1;
        rs += __shfl_xor(rs, 8, 32);
        rs += __shfl_xor(rs, 4, 32);
        rs += __shfl_xor(rs, 2, 32);
        rs += __shfl_xor(rs, 1, 32);
        const float alpha = exp2f(m[sub][i] - mn);
        l[sub][i] = l[sub][i] * alpha + rs;
        m[sub][i] = mn;
#pragma unroll
        for (int np = 0; np < 4; ++np) o[sub][np][i] *= alpha;
        // stash P in LDS to re-layout C-frag -> A-frag (row = i + 8*half)
        Plds[wave][sub][i + 8 * half][col] = (__bf16)p0;
        Plds[wave][sub][i + 8 * half][col + 16] = (__bf16)p1;
      }
    }
    asm volatile("" ::: "memory");  // keep ds_loads after ds_stores

    const bf16x16 pa0 = load_a_frag(&Plds[wave][0][0][0], 32, 0, lane);
    const bf16x16 pa1 = load_a_frag(&Plds[wave][1][0][0], 32, 0, lane);
#pragma unroll
    for (int np = 0; np < 4; ++np) {
      const bf16x16 vf = load_b_frag(Vh + (np * 16) * 2048, 2048, kb, lane);
      o[0][np] = wmma_bf16(pa0, vf, o[0][np]);
      o[1][np] = wmma_bf16(pa1, vf, o[1][np]);
    }
  }

  const int b = bh >> 4, h = bh & 15;
#pragma unroll
  for (int sub = 0; sub < 2; ++sub) {
    float linv[8];
#pragma unroll
    for (int i = 0; i < 8; ++i) linv[i] = __builtin_amdgcn_rcpf(l[sub][i]);
#pragma unroll
    for (int np = 0; np < 4; ++np) {
#pragma unroll
      for (int i = 0; i < 8; ++i) {
        const float v = o[sub][np][i] * linv[i];
        const int grow = b * 2048 + q0 + sub * 16 + i + 8 * half;
        const int gcol = h * 64 + np * 16 + col;
        ctx[grow * 1024 + gcol] = (__bf16)v;
      }
    }
  }
}

// ---------------------------------------------------------------- out GEMM
// grid (16 Ntiles, 64 Mtiles); same staged-B structure as qkv_proj_kernel.
__global__ __launch_bounds__(128) void out_proj_kernel(
    const __bf16* __restrict__ A, const __bf16* __restrict__ WT,
    const float* __restrict__ bias, float* __restrict__ out) {
  const int tid = threadIdx.x;
  const int lane = tid & 31, wave = tid >> 5;
  const int row0 = blockIdx.y * 128 + wave * 32;
  const int nbase = blockIdx.x * 64;
  const int half = lane >> 4, col = lane & 15;

  f32x8 acc[2][4];
#pragma unroll
  for (int s = 0; s < 2; ++s)
#pragma unroll
    for (int nt = 0; nt < 4; ++nt) acc[s][nt] = zero8();

  const __bf16* Arow0 = A + row0 * 1024;
  const __bf16* Arow1 = A + (row0 + 16) * 1024;

#if HAVE_ASYNC_LDS
  __shared__ __align__(16) __bf16 Bl[2][64][48];
  auto stage = [&](int k0, int buf) {
#pragma unroll
    for (int u = 0; u < 2; ++u) {
      const int c = tid + u * 128;
      const int rw = c >> 2, part = c & 3;
      async_copy16(WT + (nbase + rw) * 1024 + k0 + part * 8,
                   &Bl[buf][rw][part * 8]);
    }
  };
  stage(0, 0);
  wait_async0();
  __syncthreads();
  int buf = 0;
  for (int k0 = 0; k0 < 1024; k0 += 32) {
    if (k0 + 32 < 1024) stage(k0 + 32, buf ^ 1);
    __builtin_prefetch(Arow0 + (lane & 15) * 1024 + k0 + 128, 0, 1);
    bf16x16 a0 = load_a_frag(Arow0, 1024, k0, lane);
    bf16x16 a1 = load_a_frag(Arow1, 1024, k0, lane);
#pragma unroll
    for (int nt = 0; nt < 4; ++nt) {
      bf16x16 b = load_b_frag(&Bl[buf][nt * 16][0], 48, 0, lane);
      acc[0][nt] = wmma_bf16(a0, b, acc[0][nt]);
      acc[1][nt] = wmma_bf16(a1, b, acc[1][nt]);
    }
    wait_async0();
    __syncthreads();
    buf ^= 1;
  }
#else
#pragma unroll 2
  for (int k0 = 0; k0 < 1024; k0 += 32) {
    __builtin_prefetch(Arow0 + (lane & 15) * 1024 + k0 + 128, 0, 1);
    bf16x16 a0 = load_a_frag(Arow0, 1024, k0, lane);
    bf16x16 a1 = load_a_frag(Arow1, 1024, k0, lane);
#pragma unroll
    for (int nt = 0; nt < 4; ++nt) {
      bf16x16 b = load_b_frag(WT + (nbase + nt * 16) * 1024, 1024, k0, lane);
      acc[0][nt] = wmma_bf16(a0, b, acc[0][nt]);
      acc[1][nt] = wmma_bf16(a1, b, acc[1][nt]);
    }
  }
#endif

#pragma unroll
  for (int sub = 0; sub < 2; ++sub) {
#pragma unroll
    for (int nt = 0; nt < 4; ++nt) {
      const int gcol = nbase + nt * 16 + col;
      const float bv = bias[gcol];
#pragma unroll
      for (int i = 0; i < 8; ++i) {
        const int grow = row0 + sub * 16 + i + 8 * half;
        out[grow * 1024 + gcol] = acc[sub][nt][i] + bv;
      }
    }
  }
}

// ---------------------------------------------------------------- launcher
extern "C" void kernel_launch(void* const* d_in, const int* in_sizes, int n_in,
                              void* d_out, int out_size, void* d_ws,
                              size_t ws_size, hipStream_t stream) {
  const float* qkv   = (const float*)d_in[0];
  const float* W_in  = (const float*)d_in[1];
  const float* b_in  = (const float*)d_in[2];
  const float* W_out = (const float*)d_in[3];
  const float* b_out = (const float*)d_in[4];
  float* out = (float*)d_out;

  char* ws = (char*)d_ws;
  __bf16* qkv_bf = (__bf16*)(ws + (size_t)0);
  __bf16* WinT   = (__bf16*)(ws + (size_t)(16 << 20));
  __bf16* WoutT  = (__bf16*)(ws + (size_t)(22 << 20));
  __bf16* Qb     = (__bf16*)(ws + (size_t)(24 << 20));
  __bf16* Kb     = (__bf16*)(ws + (size_t)(32 << 20));
  __bf16* VTb    = (__bf16*)(ws + (size_t)(40 << 20));
  __bf16* ctx    = (__bf16*)(ws + (size_t)(48 << 20));

  // 1) precision conversions / weight transposes
  cvt_bf16_kernel<<<4096, 256, 0, stream>>>(qkv, qkv_bf, 8192 * 1024);
  transpose_bf16_kernel<<<6144, 256, 0, stream>>>(W_in, WinT, 3072, 3072 * 1024);
  transpose_bf16_kernel<<<2048, 256, 0, stream>>>(W_out, WoutT, 1024, 1024 * 1024);

  // 2) fused QKV projection (+bias, head-layout scatter)
  qkv_proj_kernel<<<dim3(48, 64), 128, 0, stream>>>(qkv_bf, WinT, b_in, Qb, Kb,
                                                    VTb);

  // 3) flash attention per (head, 128-query tile)
  attention_kernel<<<dim3(64, 16), 128, 0, stream>>>(Qb, Kb, VTb, ctx);

  // 4) output projection (+bias) -> f32
  out_proj_kernel<<<dim3(16, 64), 128, 0, stream>>>(ctx, WoutT, b_out, out);
}
